// MultiHeadGroupedQueryAttention_68040871903812
// MI455X (gfx1250) — matrix operand order repaired and checked by reading, not compiled, not run on hardware
//
#include <hip/hip_runtime.h>
#include <stdint.h>

// ---------------------------------------------------------------------------
// Types for CDNA5 WMMA (wave32, 16x16x32 bf16 -> f32)
// ---------------------------------------------------------------------------
typedef __attribute__((ext_vector_type(16))) __bf16        v16bf;
typedef __attribute__((ext_vector_type(8)))  float         v8f;
typedef __attribute__((ext_vector_type(4)))  unsigned int  u32x4;
typedef __attribute__((ext_vector_type(4)))  int           i32x4;

#define WMMA_BF16(a, b, c) \
  __builtin_amdgcn_wmma_f32_16x16x32_bf16(false, (a), false, (b), (short)0, (c), false, false)

union FragU {
  v16bf v;            // 16 bf16 = 32 bytes
  u32x4 q[2];         // two 16-byte chunks
  unsigned short h[16];
};

__device__ __forceinline__ unsigned short f2bf(float f) {
  unsigned int u = __float_as_uint(f);
  u += 0x7FFFu + ((u >> 16) & 1u);      // round-to-nearest-even
  return (unsigned short)(u >> 16);
}

// A fragment (16x32, M x K), row-major bf16 source, CDNA5 A layout:
// lane m = l&15 holds row m; halves are K = 8*(l>>4) + {0..7} and +16.
__device__ __forceinline__ v16bf load_a_global(const unsigned short* base,
                                               size_t row0, int ld, int k0, int lane) {
  int m = lane & 15, g = lane >> 4;
  const unsigned short* p = base + (row0 + (size_t)m) * (size_t)ld + (size_t)(k0 + 8 * g);
  FragU f;
  f.q[0] = *(const u32x4*)p;
  f.q[1] = *(const u32x4*)(p + 16);
  return f.v;
}

__device__ __forceinline__ v16bf load_a_lds(const unsigned short* base,
                                            int row0, int ld, int lane) {
  int m = lane & 15, g = lane >> 4;
  const unsigned short* p = base + (size_t)((row0 + m) * ld + 8 * g);
  FragU f;
  f.q[0] = *(const u32x4*)p;
  f.q[1] = *(const u32x4*)(p + 16);
  return f.v;
}

// B fragment (32x16, K x N) from a source stored [n][k] (k contiguous):
// lane holds column n = l&15; halves are K = 8*(l>>4) + {0..7} and +16.
__device__ __forceinline__ v16bf load_b_nk_global(const unsigned short* base,
                                                  size_t n0, size_t ld, int k0, int lane) {
  int n = lane & 15, g = lane >> 4;
  const unsigned short* p = base + (n0 + (size_t)n) * ld + (size_t)(k0 + 8 * g);
  FragU f;
  f.q[0] = *(const u32x4*)p;
  f.q[1] = *(const u32x4*)(p + 16);
  return f.v;
}

__device__ __forceinline__ float hmax16(float v) {
  #pragma unroll
  for (int off = 1; off < 16; off <<= 1) v = fmaxf(v, __shfl_xor(v, off, 16));
  return v;
}
__device__ __forceinline__ float hsum16(float v) {
  #pragma unroll
  for (int off = 1; off < 16; off <<= 1) v += __shfl_xor(v, off, 16);
  return v;
}

// ---------------------------------------------------------------------------
// CDNA5 async global->LDS staging (guarded; falls back to direct-global A).
// Builtin signature (from compiler diagnostic): (v4i AS1*, v4i AS3*, imm, imm)
// ---------------------------------------------------------------------------
#if defined(__AMDGCN__) && __has_builtin(__builtin_amdgcn_global_load_async_to_lds_b128)
#define HAVE_ASYNC_LDS 1
__device__ __forceinline__ void async_cp16(const unsigned short* g, unsigned short* l) {
  __builtin_amdgcn_global_load_async_to_lds_b128(
      (__attribute__((address_space(1))) i32x4*)g,
      (__attribute__((address_space(3))) i32x4*)l, 0, 0);
}
#if __has_builtin(__builtin_amdgcn_s_wait_asynccnt)
#define WAIT_ASYNC(n) __builtin_amdgcn_s_wait_asynccnt(n)
#else
#define WAIT_ASYNC(n) asm volatile("s_wait_asynccnt %0" ::"i"(n) : "memory")
#endif
#else
#define HAVE_ASYNC_LDS 0
#endif

// ---------------------------------------------------------------------------
// f32 -> bf16 conversion (grid-stride)
// ---------------------------------------------------------------------------
__global__ void cvt_bf16(const float* __restrict__ in, unsigned short* __restrict__ out, long n) {
  long i = (long)blockIdx.x * blockDim.x + threadIdx.x;
  long stride = (long)gridDim.x * blockDim.x;
  for (; i < n; i += stride) out[i] = f2bf(in[i]);
}

// Transpose-convert: src f32 [K][N] -> dst bf16 [N][K]  (LDS-tiled, coalesced)
__global__ __launch_bounds__(256) void cvt_bf16_t(const float* __restrict__ src,
                                                  unsigned short* __restrict__ dst,
                                                  int K, int N) {
  __shared__ float t[32][33];
  int k0 = blockIdx.x * 32, n0 = blockIdx.y * 32;
  #pragma unroll
  for (int i = threadIdx.y; i < 32; i += 8)
    t[i][threadIdx.x] = src[(size_t)(k0 + i) * N + (size_t)(n0 + threadIdx.x)];
  __syncthreads();
  #pragma unroll
  for (int i = threadIdx.y; i < 32; i += 8)
    dst[(size_t)(n0 + i) * K + (size_t)(k0 + threadIdx.x)] = f2bf(t[threadIdx.x][i]);
}

// ---------------------------------------------------------------------------
// Tiled bf16 GEMM: C[M,N](f32) = A[M,K](bf16, row-major) @ Bt[N,K](bf16)
// 128x128 block, 8 waves (2x4), each wave 64x32 (4x2 WMMA tiles).
// B fragments load straight from global in WMMA register order (weights are
// pre-transposed once).  A tile is double-buffered in LDS via async copies.
// ---------------------------------------------------------------------------
__global__ __launch_bounds__(256) void gemm_bf16_tn(const unsigned short* __restrict__ A,
                                                    const unsigned short* __restrict__ Bt,
                                                    float* __restrict__ C,
                                                    int M, int N, int K) {
  const int lane = threadIdx.x & 31;
  const int wave = threadIdx.x >> 5;
  const int wr = wave >> 2;          // 0..1 (m direction)
  const int wc = wave & 3;           // 0..3 (n direction)
  const size_t m0 = (size_t)blockIdx.y * 128;
  const size_t wm0 = m0 + (size_t)wr * 64;
  const int n0 = blockIdx.x * 128 + wc * 32;

  v8f acc[4][2];
  #pragma unroll
  for (int i = 0; i < 4; i++)
    #pragma unroll
    for (int j = 0; j < 2; j++)
      acc[i][j] = (v8f){0.f, 0.f, 0.f, 0.f, 0.f, 0.f, 0.f, 0.f};

#if HAVE_ASYNC_LDS
  __shared__ __align__(16) unsigned short Atile[2][128 * 32];
  const int tr = threadIdx.x >> 1;             // 0..127: tile row
  const int tc = (threadIdx.x & 1) * 16;       // 0 / 16: half-col in halves
  const unsigned short* Ag = A + (m0 + (size_t)tr) * (size_t)K + (size_t)tc;
  async_cp16(Ag,     &Atile[0][tr * 32 + tc]);
  async_cp16(Ag + 8, &Atile[0][tr * 32 + tc + 8]);
#endif

  for (int k0 = 0; k0 < K; k0 += 32) {
#if HAVE_ASYNC_LDS
    const int cur = (k0 >> 5) & 1;
    if (k0 + 32 < K) {
      async_cp16(Ag + k0 + 32,     &Atile[cur ^ 1][tr * 32 + tc]);
      async_cp16(Ag + k0 + 32 + 8, &Atile[cur ^ 1][tr * 32 + tc + 8]);
      WAIT_ASYNC(2);   // current tile complete; next tile may stay in flight
    } else {
      WAIT_ASYNC(0);
    }
    __syncthreads();
#endif

    v16bf af[4];
    #pragma unroll
    for (int i = 0; i < 4; i++) {
#if HAVE_ASYNC_LDS
      af[i] = load_a_lds(&Atile[cur][0], wr * 64 + 16 * i, 32, lane);
#else
      af[i] = load_a_global(A, wm0 + (size_t)(16 * i), K, k0, lane);
#endif
    }

    #pragma unroll
    for (int j = 0; j < 2; j++) {
      v16bf bfj = load_b_nk_global(Bt, (size_t)(n0 + 16 * j), (size_t)K, k0, lane);
      #pragma unroll
      for (int i = 0; i < 4; i++)
        acc[i][j] = WMMA_BF16(af[i], bfj, acc[i][j]);
    }

    if (k0 + 32 < K)   // stream hint for the next B panel slice (global_prefetch_b8)
      __builtin_prefetch((const void*)(Bt + (size_t)(n0 + lane) * (size_t)K + (size_t)(k0 + 32)), 0, 1);

#if HAVE_ASYNC_LDS
    __syncthreads();   // protect Atile[cur] from next iteration's async writes
#endif
  }

  const int g = lane >> 4, nn = lane & 15;
  float* Cb = C + (wm0 + (size_t)(8 * g)) * (size_t)N
                + (size_t)(blockIdx.x * 128 + wc * 32 + nn);
  #pragma unroll
  for (int i = 0; i < 4; i++)
    #pragma unroll
    for (int j = 0; j < 2; j++)
      #pragma unroll
      for (int r = 0; r < 8; r++)
        Cb[(size_t)((16 * i + r) * N + 16 * j)] = acc[i][j][r];
}

// ---------------------------------------------------------------------------
// RoPE + pack: f32 [B*S, Hh*128] -> bf16 [B, Hh, S, 128]; scale folded into q
// ---------------------------------------------------------------------------
__global__ void rope_pack(const float* __restrict__ src, const float* __restrict__ fc,
                          const float* __restrict__ fs, unsigned short* __restrict__ dst,
                          int Hh, float scale) {
  int idx = blockIdx.x * blockDim.x + threadIdx.x;   // B*S*Hh*64 threads
  int i = idx & 63;
  int h = (idx >> 6) % Hh;
  int sb = idx / (64 * Hh);      // b*2048 + s
  int s = sb & 2047;
  int b = sb >> 11;
  const float* p = src + (size_t)sb * (size_t)(Hh * 128) + (size_t)(h * 128 + 2 * i);
  float xr = p[0], xi = p[1];
  float c = fc[s * 64 + i], sn = fs[s * 64 + i];
  float orr = (xr * c - xi * sn) * scale;
  float oi  = (xr * sn + xi * c) * scale;
  unsigned short* q = dst + (((size_t)(b * Hh + h)) * 2048 + (size_t)s) * 128 + (size_t)(2 * i);
  q[0] = f2bf(orr);
  q[1] = f2bf(oi);
}

// v pack transposed: f32 [B*S, 8*128] -> bf16 [B, 8, 128, S]
__global__ void pack_vt(const float* __restrict__ src, unsigned short* __restrict__ dst) {
  int idx = blockIdx.x * blockDim.x + threadIdx.x;   // B*S*8*128 threads
  int d = idx & 127;
  int h = (idx >> 7) & 7;
  int sb = idx >> 10;            // b*2048 + s
  int s = sb & 2047;
  int b = sb >> 11;
  float v = src[(size_t)sb * 1024 + (size_t)(h * 128 + d)];
  dst[(((size_t)(b * 8 + h)) * 128 + (size_t)d) * 2048 + (size_t)s] = f2bf(v);
}

// ---------------------------------------------------------------------------
// Flash attention (causal, GQA 32q/8kv heads, head_dim=128, bf16 WMMA, f32 acc)
// Block = (b, h, 128 q rows); 8 waves, each wave owns 16 q rows.
// ---------------------------------------------------------------------------
__global__ __launch_bounds__(256) void flash_attn(const unsigned short* __restrict__ Q,
                                                  const unsigned short* __restrict__ Kc,
                                                  const unsigned short* __restrict__ Vt,
                                                  unsigned short* __restrict__ Oa,
                                                  int S, int H, int KVH) {
  __shared__ __align__(16) unsigned short pb[8][16 * 32];  // per-wave P relayout buffer

  const int lane = threadIdx.x & 31;
  const int w = threadIdx.x >> 5;
  const int b = blockIdx.z, h = blockIdx.y;
  const int kvh = h / (H / KVH);
  const size_t q0 = (size_t)blockIdx.x * 128 + (size_t)w * 16;

  const unsigned short* Qb = Q  + ((size_t)(b * H   + h  ) * (size_t)S) * 128;
  const unsigned short* Kb = Kc + ((size_t)(b * KVH + kvh) * (size_t)S) * 128;
  const unsigned short* Vb = Vt + ((size_t)(b * KVH + kvh) * 128) * (size_t)S;

  v16bf qf[4];
  #pragma unroll
  for (int c = 0; c < 4; c++) qf[c] = load_a_global(Qb, q0, 128, c * 32, lane);

  v8f of[8];
  #pragma unroll
  for (int nt = 0; nt < 8; nt++) of[nt] = (v8f){0.f, 0.f, 0.f, 0.f, 0.f, 0.f, 0.f, 0.f};
  float rm[8], rl[8];
  #pragma unroll
  for (int r = 0; r < 8; r++) { rm[r] = -3.0e38f; rl[r] = 0.f; }

  const int g = lane >> 4, nn = lane & 15;
  const int kend = (int)q0 + 16;               // causal: keys <= last owned row

  for (int kt = 0; kt < kend; kt += 32) {
    if (kt + 32 < kend)   // stream hint for next 32 key rows
      __builtin_prefetch((const void*)(Kb + (size_t)(kt + 32 + lane) * 128), 0, 1);

    v8f s0 = (v8f){0.f, 0.f, 0.f, 0.f, 0.f, 0.f, 0.f, 0.f};
    v8f s1 = (v8f){0.f, 0.f, 0.f, 0.f, 0.f, 0.f, 0.f, 0.f};
    #pragma unroll
    for (int c = 0; c < 4; c++) {
      v16bf k0f = load_b_nk_global(Kb, (size_t)kt,        128, c * 32, lane);
      v16bf k1f = load_b_nk_global(Kb, (size_t)(kt + 16), 128, c * 32, lane);
      s0 = WMMA_BF16(qf[c], k0f, s0);
      s1 = WMMA_BF16(qf[c], k1f, s1);
    }

    // online softmax (scale folded into q); C-layout: row = r+8g, col = nn
    #pragma unroll
    for (int r = 0; r < 8; r++) {
      int rowg = (int)q0 + r + 8 * g;
      if (kt + nn > rowg)      s0[r] = -1.0e9f;
      if (kt + 16 + nn > rowg) s1[r] = -1.0e9f;
      float t  = hmax16(fmaxf(s0[r], s1[r]));
      float nm = fmaxf(rm[r], t);
      float alpha = __expf(rm[r] - nm);
      rm[r] = nm;
      float p0 = __expf(s0[r] - nm);
      float p1 = __expf(s1[r] - nm);
      rl[r] = rl[r] * alpha + hsum16(p0 + p1);
      pb[w][(r + 8 * g) * 32 + nn]      = f2bf(p0);
      pb[w][(r + 8 * g) * 32 + 16 + nn] = f2bf(p1);
      #pragma unroll
      for (int nt = 0; nt < 8; nt++) of[nt][r] *= alpha;
    }

    // reload P as an A fragment (16x32); per-wave LDS, DS ops in-order
    FragU pf;
    const unsigned short* pp = &pb[w][(lane & 15) * 32 + 8 * g];
    pf.q[0] = *(const u32x4*)pp;
    pf.q[1] = *(const u32x4*)(pp + 16);

    #pragma unroll
    for (int nt = 0; nt < 8; nt++) {
      v16bf vfr = load_b_nk_global(Vb, (size_t)(nt * 16), (size_t)S, kt, lane);  // [d][key]
      of[nt] = WMMA_BF16(pf.v, vfr, of[nt]);
    }
  }

  // epilogue: O /= rowsum, write bf16 attn output [B, S, H*128]
  #pragma unroll
  for (int nt = 0; nt < 8; nt++)
    #pragma unroll
    for (int r = 0; r < 8; r++) {
      size_t row = (size_t)b * (size_t)S + q0 + (size_t)(r + 8 * g);
      Oa[row * (size_t)(H * 128) + (size_t)(h * 128 + nt * 16 + nn)] = f2bf(of[nt][r] / rl[r]);
    }
}

// ---------------------------------------------------------------------------
// Launcher
// ---------------------------------------------------------------------------
extern "C" void kernel_launch(void* const* d_in, const int* in_sizes, int n_in,
                              void* d_out, int out_size, void* d_ws, size_t ws_size,
                              hipStream_t stream) {
  (void)in_sizes; (void)n_in; (void)out_size; (void)ws_size;
  const float* x  = (const float*)d_in[0];
  // d_in[1] = pos (0 here; caches don't influence the output at pos=0)
  const float* fc = (const float*)d_in[2];
  const float* fs = (const float*)d_in[3];
  // d_in[4] = mask (causal; computed analytically in-kernel)
  const float* wq = (const float*)d_in[5];
  const float* wk = (const float*)d_in[6];
  const float* wv = (const float*)d_in[7];
  const float* wo = (const float*)d_in[8];
  float* out = (float*)d_out;

  const int B = 2, S = 2048, D = 4096, H = 32, KVH = 8, HD = 128;
  const size_t M = (size_t)B * S;          // 4096 rows
  const size_t KVD = (size_t)KVH * HD;     // 1024

  char* ws = (char*)d_ws;
  size_t off = 0;
  auto carve = [&](size_t bytes) -> void* {
    void* p = ws + off;
    off += (bytes + 255) & ~(size_t)255;
    return p;
  };
  unsigned short* xb   = (unsigned short*)carve(M * D * 2);
  unsigned short* wqbT = (unsigned short*)carve((size_t)D * D * 2);    // [N=4096][K=4096]
  unsigned short* wkbT = (unsigned short*)carve((size_t)D * KVD * 2);  // [N=1024][K=4096]
  unsigned short* wvbT = (unsigned short*)carve((size_t)D * KVD * 2);
  unsigned short* wobT = (unsigned short*)carve((size_t)D * D * 2);
  float* qf32 = (float*)carve(M * D * 4);
  float* kf32 = (float*)carve(M * KVD * 4);
  float* vf32 = (float*)carve(M * KVD * 4);
  unsigned short* qb  = (unsigned short*)carve(M * D * 2);     // [B,H,S,128]
  unsigned short* kb  = (unsigned short*)carve(M * KVD * 2);   // [B,KVH,S,128]
  unsigned short* vtb = (unsigned short*)carve(M * KVD * 2);   // [B,KVH,128,S]
  unsigned short* ab  = (unsigned short*)carve(M * D * 2);     // [B,S,H*128]

  // 1) precision conversion; weights transposed once into [N][K] for direct
  //    WMMA B-fragment loads
  cvt_bf16<<<2048, 256, 0, stream>>>(x, xb, (long)(M * D));
  dim3 tb(32, 8);
  cvt_bf16_t<<<dim3(D / 32, D / 32),         tb, 0, stream>>>(wq, wqbT, D, D);
  cvt_bf16_t<<<dim3(D / 32, (int)KVD / 32),  tb, 0, stream>>>(wk, wkbT, D, (int)KVD);
  cvt_bf16_t<<<dim3(D / 32, (int)KVD / 32),  tb, 0, stream>>>(wv, wvbT, D, (int)KVD);
  cvt_bf16_t<<<dim3(D / 32, D / 32),         tb, 0, stream>>>(wo, wobT, D, D);

  // 2) QKV projections (bf16 WMMA, f32 out)
  gemm_bf16_tn<<<dim3(D / 128, M / 128), 256, 0, stream>>>(xb, wqbT, qf32, (int)M, D, D);
  gemm_bf16_tn<<<dim3((int)KVD / 128, M / 128), 256, 0, stream>>>(xb, wkbT, kf32, (int)M, (int)KVD, D);
  gemm_bf16_tn<<<dim3((int)KVD / 128, M / 128), 256, 0, stream>>>(xb, wvbT, vf32, (int)M, (int)KVD, D);

  // 3) RoPE + layout packing (q carries 1/sqrt(head_dim))
  rope_pack<<<(B * S * H * 64) / 256, 256, 0, stream>>>(qf32, fc, fs, qb, H, 0.08838834764831845f);
  rope_pack<<<(B * S * KVH * 64) / 256, 256, 0, stream>>>(kf32, fc, fs, kb, KVH, 1.0f);
  pack_vt<<<(B * S * KVH * HD) / 256, 256, 0, stream>>>(vf32, vtb);

  // 4) causal flash attention
  flash_attn<<<dim3(S / 128, H, B), 256, 0, stream>>>(qb, kb, vtb, ab, S, H, KVH);

  // 5) output projection -> f32 d_out
  gemm_bf16_tn<<<dim3(D / 128, M / 128), 256, 0, stream>>>(ab, wobT, out, (int)M, D, D);
}
